// DiversifiedSelfAttention_81638738362593
// MI455X (gfx1250) — matrix operand order repaired and checked by reading, not compiled, not run on hardware
//
#include <hip/hip_runtime.h>
#include <hip/hip_bf16.h>

// ---------------------------------------------------------------------------
// Shapes (fixed by the reference)
// ---------------------------------------------------------------------------
static constexpr int BB  = 2;
static constexpr int LL  = 2048;
static constexpr int DD  = 1024;
static constexpr int HH  = 16;
static constexpr int HD  = 64;
static constexpr int MM  = BB * LL;          // 4096 rows for the projections

typedef __attribute__((ext_vector_type(16))) __bf16 v16bf;
typedef __attribute__((ext_vector_type(8)))  float  v8f;
typedef __attribute__((ext_vector_type(4)))  float  f32x4;   // native vector (NT-store OK)

union FragBF {               // one WMMA 16-bf16 operand per lane = 2 x 16B LDS loads
  uint4 u[2];
  v16bf v;
};

__device__ __forceinline__ v8f wmma_bf16(const FragBF& a, const FragBF& b, v8f c) {
  return __builtin_amdgcn_wmma_f32_16x16x32_bf16(false, a.v, false, b.v,
                                                 (short)0, c, false, false);
}

// ---------------------------------------------------------------------------
// fp32 -> bf16 convert
// ---------------------------------------------------------------------------
__global__ void cvt_bf16_kernel(const float* __restrict__ in, __bf16* __restrict__ out, int n) {
  for (int i = blockIdx.x * blockDim.x + threadIdx.x; i < n; i += gridDim.x * blockDim.x)
    out[i] = (__bf16)in[i];
}

// ---------------------------------------------------------------------------
// Tiled bf16 GEMM:  C[M,N] = A[M,K] @ W[K,N] + bias
//   EPI==0 : convert to bf16 and scatter to [B][H][L][HD]   (QKV projections)
//   EPI==1 : store fp32 dense [M,N]                         (output projection)
// Block: 128 threads (4 waves), 64x64 output tile, K step 32.
// ---------------------------------------------------------------------------
template <int EPI>
__global__ __launch_bounds__(128) void gemm_kernel(
    const __bf16* __restrict__ A, const __bf16* __restrict__ W,
    const float* __restrict__ bias, void* __restrict__ Cout,
    int M, int N, int K) {
  constexpr int TM = 64, TN = 64, TK = 32;
  __shared__ __align__(16) __bf16 As[TM * TK];   // [row][k]
  __shared__ __align__(16) __bf16 BsT[TN * TK];  // [n][k]  (transposed on store)

  const int tid  = threadIdx.x;
  const int lane = tid & 31;
  const int wid  = tid >> 5;
  const int mBase = blockIdx.y * TM;
  const int nBase = blockIdx.x * TN;
  const int rowLo = (lane >> 4) << 3;   // 0 or 8

  v8f acc[4] = {};

  for (int kk = 0; kk < K; kk += TK) {
    __syncthreads();
    // A tile: 64x32 bf16, coalesced 16B loads, row-major in LDS
    for (int i = tid; i < (TM * TK / 8); i += 128) {
      int e = i * 8, r = e >> 5, c = e & 31;
      *(uint4*)&As[e] = *(const uint4*)&A[(size_t)(mBase + r) * K + kk + c];
    }
    // W tile: 32x64 bf16, stored transposed so B fragments are contiguous
    for (int i = tid; i < (TK * TN / 8); i += 128) {
      int e = i * 8, r = e >> 6, c = e & 63;
      uint4 w = *(const uint4*)&W[(size_t)(kk + r) * N + nBase + c];
      const __bf16* w8 = (const __bf16*)&w;
#pragma unroll
      for (int j = 0; j < 8; ++j) BsT[(c + j) * TK + r] = w8[j];
    }
    __syncthreads();

    // issue ALL fragment loads first, then the WMMA group (overlap DS latency)
    FragBF a, b[4];
    const int arow = wid * 16 + (lane & 15);
    const int kc = (lane >> 4) << 4;
    a.u[0] = *(const uint4*)&As[arow * TK + 0 + kc];
    a.u[1] = *(const uint4*)&As[arow * TK + 8 + kc];
#pragma unroll
    for (int n = 0; n < 4; ++n) {
      const int bcol = n * 16 + (lane & 15);
      b[n].u[0] = *(const uint4*)&BsT[bcol * TK + kc + 0];
      b[n].u[1] = *(const uint4*)&BsT[bcol * TK + kc + 8];
    }
#pragma unroll
    for (int n = 0; n < 4; ++n) acc[n] = wmma_bf16(a, b[n], acc[n]);
  }

  // Epilogue: C/D layout -> row = vgpr + (lane>=16)*8, col = lane&15
#pragma unroll
  for (int n = 0; n < 4; ++n) {
    const int ng = nBase + n * 16 + (lane & 15);
    const float bval = bias[ng];
#pragma unroll
    for (int r = 0; r < 8; ++r) {
      const int mg = mBase + wid * 16 + rowLo + r;
      const float v = acc[n][r] + bval;
      if (EPI == 0) {
        const int b = mg >> 11, l = mg & (LL - 1);
        const int h = ng >> 6,  hd = ng & (HD - 1);
        ((__bf16*)Cout)[(((size_t)b * HH + h) * LL + l) * HD + hd] = (__bf16)v;
      } else {
        ((float*)Cout)[(size_t)mg * N + ng] = v;
      }
    }
  }
}

// ---------------------------------------------------------------------------
// Fused causal attention.
// One block = (b, h, 64 query rows), 128 threads (4 waves, 16 q-rows each).
// Pass A (specialized loop): scores via WMMA -> rowsum
//        (no running max needed: scores clamped to [-50,50] by the reference).
// Pass B (specialized loop): recompute scores, write normalized attn
//        (non-temporal), O += P@V via second WMMA.
// Causal mask folded into the clamp cap: cap = (j<=i) ? 50 : -50, then
// sv = fmin(fmax(s,-50), cap)  ==> masked entries become exactly -50,
// and the select is a constant-constant v_cndmask (no exec-mask blocks).
// Fully-masked tail columns get the closed-form exp(-50)/rowsum fill.
// ---------------------------------------------------------------------------
__global__ __launch_bounds__(128) void attn_kernel(
    const __bf16* __restrict__ Qh, const __bf16* __restrict__ Kh,
    const __bf16* __restrict__ Vh, float* __restrict__ attnOut,
    __bf16* __restrict__ Oh) {
  __shared__ __align__(16) __bf16 qt[64 * HD];      // Q tile  [qrow][hd]
  __shared__ __align__(16) __bf16 kt[32 * HD];      // K chunk [key][hd]
  __shared__ __align__(16) __bf16 vtT[HD * 32];     // V chunk [hd][key] (transposed)
  __shared__ __align__(16) __bf16 pbuf[4][16 * 32]; // per-wave P (accum->Afrag relayout)
  __shared__ float invbuf[64];

  const int tid = threadIdx.x, lane = tid & 31, wid = tid >> 5;
  const int qb = blockIdx.x, h = blockIdx.y, b = blockIdx.z;
  const int q0 = qb * 64;
  const size_t headOff = ((size_t)b * HH + h) * (size_t)LL * HD;
  const __bf16* Qp = Qh + headOff + (size_t)q0 * HD;
  const __bf16* Kp = Kh + headOff;
  const __bf16* Vp = Vh + headOff;
  float* attnBase = attnOut + ((size_t)b * HH + h) * (size_t)LL * LL;

  for (int i = tid; i < 64 * HD / 8; i += 128)
    *(uint4*)&qt[i * 8] = *(const uint4*)&Qp[i * 8];
  __syncthreads();

  const int numC = (q0 + 64) / 32;                 // causal: chunks we compute
  const float expm50 = 1.928749847963918e-22f;     // exp(-50)
  const int rowLo = (lane >> 4) << 3;              // 0 or 8
  const int myRow0 = wid * 16 + rowLo;             // local base row of my accum rows

  // ---- Q A-fragments are chunk-invariant: load once
  FragBF a0, a1;
  {
    const int arow = wid * 16 + (lane & 15);
    a0.u[0] = *(const uint4*)&qt[arow * HD + 0 + 2 * rowLo];
    a0.u[1] = *(const uint4*)&qt[arow * HD + 8 + 2 * rowLo];
    a1.u[0] = *(const uint4*)&qt[arow * HD + 32 + 2 * rowLo];
    a1.u[1] = *(const uint4*)&qt[arow * HD + 40 + 2 * rowLo];
  }

  float rsum[8] = {};
  float inv[8];
  v8f oacc[4] = {};

  // =========================== PASS A: row sums ===========================
  for (int ch = 0; ch < numC; ++ch) {
    const int kBase = ch * 32;
    __syncthreads();
    for (int i = tid; i < 32 * HD / 8; i += 128)
      *(uint4*)&kt[i * 8] = *(const uint4*)&Kp[(size_t)kBase * HD + i * 8];
    __syncthreads();
    if (ch + 1 < numC)                               // prefetch next K chunk
      __builtin_prefetch(&Kp[(size_t)(kBase + 32) * HD + tid * 16], 0, 3);

    // all 4 score B-fragments first, then the WMMA group
    FragBF bf[4];
    const int kc = (lane >> 4) << 4;
#pragma unroll
    for (int n = 0; n < 2; ++n) {
      const int bcol = n * 16 + (lane & 15);         // key within chunk
      bf[2 * n + 0].u[0] = *(const uint4*)&kt[bcol * HD + 0 + kc];
      bf[2 * n + 0].u[1] = *(const uint4*)&kt[bcol * HD + 8 + kc];
      bf[2 * n + 1].u[0] = *(const uint4*)&kt[bcol * HD + 32 + kc];
      bf[2 * n + 1].u[1] = *(const uint4*)&kt[bcol * HD + 40 + kc];
    }
    v8f s[2] = {};
    s[0] = wmma_bf16(a0, bf[0], s[0]);
    s[0] = wmma_bf16(a1, bf[1], s[0]);
    s[1] = wmma_bf16(a0, bf[2], s[1]);
    s[1] = wmma_bf16(a1, bf[3], s[1]);

#pragma unroll
    for (int n = 0; n < 2; ++n)
#pragma unroll
      for (int r = 0; r < 8; ++r) {
        const int gi = q0 + myRow0 + r;
        const int gj = kBase + n * 16 + (lane & 15);
        const float cap = (gj <= gi) ? 50.0f : -50.0f;       // v_cndmask
        const float sv = fminf(fmaxf(s[n][r] * 0.125f, -50.0f), cap);
        rsum[r] += __expf(sv);
      }
  }

  // rowsum += fully-masked tail (reference keeps exp(-50) terms);
  // reduce across the 16 lanes that share each row set
  {
    const float tail = (float)(LL - numC * 32) * expm50;
#pragma unroll
    for (int r = 0; r < 8; ++r) {
      float v = rsum[r] + tail;
      v += __shfl_xor(v, 1);
      v += __shfl_xor(v, 2);
      v += __shfl_xor(v, 4);
      v += __shfl_xor(v, 8);
      inv[r] = 1.0f / v;
    }
    if ((lane & 15) == 0)
#pragma unroll
      for (int r = 0; r < 8; ++r) invbuf[myRow0 + r] = inv[r];
  }

  // ================= PASS B: attn writes + O accumulation =================
  for (int ch = 0; ch < numC; ++ch) {
    const int kBase = ch * 32;
    __syncthreads();
    for (int i = tid; i < 32 * HD / 8; i += 128)
      *(uint4*)&kt[i * 8] = *(const uint4*)&Kp[(size_t)kBase * HD + i * 8];
    for (int i = tid; i < 32 * HD / 8; i += 128) {
      int e = i * 8, kr = e >> 6, c = e & 63;
      uint4 w = *(const uint4*)&Vp[(size_t)(kBase + kr) * HD + c];
      const __bf16* w8 = (const __bf16*)&w;
#pragma unroll
      for (int j = 0; j < 8; ++j) vtT[(c + j) * 32 + kr] = w8[j];
    }
    __syncthreads();
    if (ch + 1 < numC) {                             // prefetch next K+V chunks
      __builtin_prefetch(&Kp[(size_t)(kBase + 32) * HD + tid * 16], 0, 3);
      __builtin_prefetch(&Vp[(size_t)(kBase + 32) * HD + tid * 16], 0, 3);
    }

    FragBF bf[4];
    const int kc = (lane >> 4) << 4;
#pragma unroll
    for (int n = 0; n < 2; ++n) {
      const int bcol = n * 16 + (lane & 15);
      bf[2 * n + 0].u[0] = *(const uint4*)&kt[bcol * HD + 0 + kc];
      bf[2 * n + 0].u[1] = *(const uint4*)&kt[bcol * HD + 8 + kc];
      bf[2 * n + 1].u[0] = *(const uint4*)&kt[bcol * HD + 32 + kc];
      bf[2 * n + 1].u[1] = *(const uint4*)&kt[bcol * HD + 40 + kc];
    }
    v8f s[2] = {};
    s[0] = wmma_bf16(a0, bf[0], s[0]);
    s[0] = wmma_bf16(a1, bf[1], s[0]);
    s[1] = wmma_bf16(a0, bf[2], s[1]);
    s[1] = wmma_bf16(a1, bf[3], s[1]);

#pragma unroll
    for (int n = 0; n < 2; ++n)
#pragma unroll
      for (int r = 0; r < 8; ++r) {
        const int gi = q0 + myRow0 + r;
        const int gj = kBase + n * 16 + (lane & 15);
        const float cap = (gj <= gi) ? 50.0f : -50.0f;       // v_cndmask
        const float sv = fminf(fmaxf(s[n][r] * 0.125f, -50.0f), cap);
        const float pa = __expf(sv) * inv[r];
        __builtin_nontemporal_store(pa, &attnBase[(size_t)gi * LL + gj]);
        pbuf[wid][(rowLo + r) * 32 + n * 16 + (lane & 15)] = (__bf16)pa;
      }

    // ---- O(16x64) += P(16x32) @ V(32x64); LDS ops are in-order per wave
    {
      FragBF pa, vb[4];
      const int prow = lane & 15;
      pa.u[0] = *(const uint4*)&pbuf[wid][prow * 32 + 0 + 2 * rowLo];
      pa.u[1] = *(const uint4*)&pbuf[wid][prow * 32 + 8 + 2 * rowLo];
      const int kc2 = (lane >> 4) << 4;              // key 0 or 16
#pragma unroll
      for (int n = 0; n < 4; ++n) {
        const int vcol = n * 16 + (lane & 15);       // hd
        vb[n].u[0] = *(const uint4*)&vtT[vcol * 32 + kc2 + 0];
        vb[n].u[1] = *(const uint4*)&vtT[vcol * 32 + kc2 + 8];
      }
#pragma unroll
      for (int n = 0; n < 4; ++n) oacc[n] = wmma_bf16(pa, vb[n], oacc[n]);
    }
  }

  // store O as bf16 into [B][L][D] for the output projection
#pragma unroll
  for (int n = 0; n < 4; ++n)
#pragma unroll
    for (int r = 0; r < 8; ++r) {
      const int lrow = q0 + myRow0 + r;
      const int hd = n * 16 + (lane & 15);
      Oh[((size_t)b * LL + lrow) * DD + h * HD + hd] = (__bf16)oacc[n][r];
    }

  __syncthreads();
  // fill fully-masked tail columns: attn = exp(-50) / rowsum (vectorized, NT)
  const int kStart = numC * 32;
  for (int r = 0; r < 64; ++r) {
    const float v = expm50 * invbuf[r];
    const f32x4 vv = {v, v, v, v};
    f32x4* rowp = (f32x4*)(attnBase + (size_t)(q0 + r) * LL);
    for (int c4 = kStart / 4 + tid; c4 < LL / 4; c4 += 128)
      __builtin_nontemporal_store(vv, &rowp[c4]);
  }
}

// ---------------------------------------------------------------------------
// Launch
// ---------------------------------------------------------------------------
extern "C" void kernel_launch(void* const* d_in, const int* in_sizes, int n_in,
                              void* d_out, int out_size, void* d_ws, size_t ws_size,
                              hipStream_t stream) {
  const float* x  = (const float*)d_in[0];
  const float* wq = (const float*)d_in[1];
  const float* bq = (const float*)d_in[2];
  const float* wk = (const float*)d_in[3];
  const float* bk = (const float*)d_in[4];
  const float* wv = (const float*)d_in[5];
  const float* bv = (const float*)d_in[6];
  const float* wo = (const float*)d_in[7];
  const float* bo = (const float*)d_in[8];

  float* outp  = (float*)d_out;                       // [B,L,D]
  float* attnp = outp + (size_t)BB * LL * DD;         // [B,H,L,L]

  // workspace layout (bytes), total 48 MB
  char* ws = (char*)d_ws;
  const size_t NX = (size_t)MM * DD;                  // 4M elements
  const size_t NW = (size_t)DD * DD;                  // 1M elements
  __bf16* xb  = (__bf16*)(ws);
  __bf16* wqb = (__bf16*)(ws + NX * 2);
  __bf16* wkb = (__bf16*)(ws + NX * 2 + NW * 2);
  __bf16* wvb = (__bf16*)(ws + NX * 2 + NW * 4);
  __bf16* wob = (__bf16*)(ws + NX * 2 + NW * 6);
  __bf16* Qb  = (__bf16*)(ws + NX * 2 + NW * 8);
  __bf16* Kb  = (__bf16*)(ws + NX * 4 + NW * 8);
  __bf16* Vb  = (__bf16*)(ws + NX * 6 + NW * 8);
  __bf16* Ob  = (__bf16*)(ws + NX * 8 + NW * 8);

  // 1) converts
  cvt_bf16_kernel<<<4096, 256, 0, stream>>>(x, xb, (int)NX);
  cvt_bf16_kernel<<<1024, 256, 0, stream>>>(wq, wqb, (int)NW);
  cvt_bf16_kernel<<<1024, 256, 0, stream>>>(wk, wkb, (int)NW);
  cvt_bf16_kernel<<<1024, 256, 0, stream>>>(wv, wvb, (int)NW);
  cvt_bf16_kernel<<<1024, 256, 0, stream>>>(wo, wob, (int)NW);

  // 2) QKV projections (scatter to [B][H][L][HD] bf16)
  dim3 gg(DD / 64, MM / 64);
  gemm_kernel<0><<<gg, 128, 0, stream>>>(xb, wqb, bq, Qb, MM, DD, DD);
  gemm_kernel<0><<<gg, 128, 0, stream>>>(xb, wkb, bk, Kb, MM, DD, DD);
  gemm_kernel<0><<<gg, 128, 0, stream>>>(xb, wvb, bv, Vb, MM, DD, DD);

  // 3) fused attention (+ full attn output)
  dim3 ga(LL / 64, HH, BB);
  attn_kernel<<<ga, 128, 0, stream>>>(Qb, Kb, Vb, attnp, Ob);

  // 4) output projection (fp32 to d_out)
  gemm_kernel<1><<<gg, 128, 0, stream>>>(Ob, wob, bo, outp, MM, DD, DD);
}
